// Flow_68015102099541
// MI455X (gfx1250) — compile-verified, matrix operand order checked by reading, not run on hardware
//
#include <hip/hip_runtime.h>
#include <hip/hip_bf16.h>
#include <cstdint>

typedef __attribute__((ext_vector_type(16))) _Float16 v16h;
typedef __attribute__((ext_vector_type(8)))  _Float16 v8h;
typedef __attribute__((ext_vector_type(8)))  float    v8f;
typedef __attribute__((ext_vector_type(4)))  float    v4f;

#define B_    128
#define CIN_  64
#define NF_   16
#define COUT_ 256
#define H_    20
#define W_    20
#define HW_   (H_*W_)          // 400
#define M_    (B_*HW_)         // 51200
#define K_    (CIN_*NF_)       // 1024  (permuted order: k' = nf*64 + cin)
#define N_    COUT_            // 256

#define TM_   64               // M rows per block
#define KC_   128              // K chunk = 2 flows x 64 cin
#define NCHUNK_ (K_/KC_)       // 8
#define LROW_ 136              // LDS A-slab row stride in halves (128 + 8 pad)
#define TROW_ 20               // epilogue transpose row stride in floats (16 + 4)

struct Tap { int off[4]; float w[4]; };

// ---------------------------------------------------------------------------
// Kernel 0: per-(nf,h,w) bilinear tap table. Offsets into the 20x20 plane,
// -1 => zero tap. Replicates the reference's w10 = s*(1-t) quirk.
// ---------------------------------------------------------------------------
__global__ __launch_bounds__(256) void flow_table_kernel(const float* __restrict__ flow,
                                                         Tap* __restrict__ table) {
  int t = blockIdx.x * blockDim.x + threadIdx.x;
  if (t >= NF_ * HW_) return;
  int nf = t / HW_;
  int hw = t - nf * HW_;
  int h  = hw / W_;
  int w  = hw - h * W_;
  const float* fp = flow + (((size_t)nf * H_ + h) * W_ + w) * 2;
  float ix = (float)h + fp[0];
  float iy = (float)w + fp[1];
  float bxf = floorf(ix), byf = floorf(iy);
  float s  = ix - bxf;
  float tt = iy - byf;
  int bx = (int)bxf, by = (int)byf;
  float ws[4] = { (1.f - s) * (1.f - tt),
                  s * (1.f - tt),
                  s * (1.f - tt),          // reference bug kept on purpose
                  s * tt };
  int xs[4] = { bx, bx + 1, bx,     bx + 1 };
  int ys[4] = { by, by,     by + 1, by + 1 };
  Tap e;
#pragma unroll
  for (int j = 0; j < 4; ++j) {
    int cx = min(max(xs[j], 0), H_);
    int cy = min(max(ys[j], 0), W_);
    e.off[j] = (cx == H_ || cy == W_) ? -1 : (cx * W_ + cy);
    e.w[j]   = ws[j];
  }
  table[(size_t)hw * NF_ + nf] = e;
}

// ---------------------------------------------------------------------------
// Kernel 1: x (B,CIN,H,W) fp32 -> xt (B,HW,CIN) fp16. Channel-contiguous so a
// single b128 load fetches one bilinear tap for 8 channels.
// ---------------------------------------------------------------------------
__global__ __launch_bounds__(256) void prep_xt_kernel(const float* __restrict__ x,
                                                      _Float16* __restrict__ xt) {
  int t = blockIdx.x * blockDim.x + threadIdx.x;       // one (b,hw) per thread
  if (t >= M_) return;
  int b  = t / HW_;
  int hw = t - b * HW_;
  const float* xp = x + (size_t)b * CIN_ * HW_ + hw;   // stride HW_ over cin
  _Float16* dst = xt + (size_t)t * CIN_;
#pragma unroll
  for (int g = 0; g < CIN_ / 8; ++g) {
    v8h v;
#pragma unroll
    for (int j = 0; j < 8; ++j) v[j] = (_Float16)xp[(size_t)(g * 8 + j) * HW_];
    *(v8h*)(dst + g * 8) = v;
  }
}

// ---------------------------------------------------------------------------
// Kernel 2: combinator fp32 (N, cin*16+nf) -> fp16 (N, nf*64+cin).
// Shared K-permutation of A and B leaves the GEMM result unchanged.
// ---------------------------------------------------------------------------
__global__ __launch_bounds__(256) void cvt_comb_kernel(const float* __restrict__ c,
                                                       _Float16* __restrict__ cp) {
  int i = blockIdx.x * blockDim.x + threadIdx.x;
  if (i >= N_ * K_) return;
  int n   = i >> 10;
  int kp  = i & 1023;
  int nf  = kp >> 6;
  int cin = kp & 63;
  cp[i] = (_Float16)c[((size_t)n << 10) + cin * NF_ + nf];
}

// ---------------------------------------------------------------------------
// Producer: bilinear warp of one K-chunk (2 flows x 64 cin) into an LDS slab.
// ---------------------------------------------------------------------------
__device__ __forceinline__ void produce_chunk(int c, _Float16* buf, int row, int tgrp,
                                              const Tap* tp, const _Float16* xb) {
#pragma unroll
  for (int i = 0; i < 4; ++i) {
    int tau  = tgrp + (i << 2);                        // 0..15
    int nfl  = tau >> 3;                               // 0..1
    int cin0 = (tau & 7) * 8;
    Tap e = tp[c * 2 + nfl];
    v8h a = {};
#pragma unroll
    for (int j = 0; j < 4; ++j) {
      if (e.off[j] >= 0) {
        v8h tv = *(const v8h*)(xb + (size_t)e.off[j] * CIN_ + cin0);
        _Float16 wh = (_Float16)e.w[j];
        v8h wv = {wh, wh, wh, wh, wh, wh, wh, wh};
        a += tv * wv;                                  // v_pk_fma_f16
      }
    }
    *(v8h*)(buf + (size_t)row * LROW_ + nfl * 64 + cin0) = a;
  }
}

// ---------------------------------------------------------------------------
// Kernel 3 (fused): warp -> double-buffered LDS A-slab -> WMMA GEMM ->
// per-wave LDS transpose -> coalesced (B,COUT,H,W) stores (+bias).
// Block: 256 threads = 8 waves. Tile 64M x 256N, K in 8 chunks of 128.
// Wave w: rows (w&3)*16.., N half (w>>2)*128.. (8 subtiles, 64 acc VGPRs).
// ---------------------------------------------------------------------------
__global__ __launch_bounds__(256) void fused_warp_gemm_kernel(
    const _Float16* __restrict__ xt, const Tap* __restrict__ table,
    const _Float16* __restrict__ Bm, const float* __restrict__ bias,
    float* __restrict__ out) {
  __shared__ __align__(16) _Float16 lds[2][TM_ * LROW_];   // 2 x 17,408 B

  const int mBase = blockIdx.x * TM_;

  // ---- phase-1 (producer) mapping: one row per thread, 4 tasks/chunk ----
  const int row  = threadIdx.x & 63;
  const int tgrp = threadIdx.x >> 6;                   // 0..3
  const int m1   = mBase + row;
  const int b1   = m1 / HW_;
  const int hw1  = m1 - b1 * HW_;
  const Tap* tp  = table + (size_t)hw1 * NF_;
  const _Float16* xb = xt + (size_t)b1 * HW_ * CIN_;

  // ---- phase-2 (consumer) mapping ----
  const int lane  = threadIdx.x & 31;
  const int wave  = threadIdx.x >> 5;
  const int l16   = lane & 15;
  const int hi16  = lane >> 4;
  const int khalf = hi16 * 8;
  const int r0    = (wave & 3) * 16;
  const int nh    = wave >> 2;                          // 0..1 -> N half
  const _Float16* bBase = Bm + (size_t)(nh * 128 + l16) * K_ + khalf;

  v8f acc[8] = {};

  // prologue: fill buffer 0, then overlap produce(c+1) with consume(c)
  produce_chunk(0, lds[0], row, tgrp, tp, xb);
  __syncthreads();

  for (int c = 0; c < NCHUNK_; ++c) {
    if (c + 1 < NCHUNK_)
      produce_chunk(c + 1, lds[(c + 1) & 1], row, tgrp, tp, xb);

    const _Float16* aLds = lds[c & 1] + (size_t)(r0 + l16) * LROW_ + khalf;
#pragma unroll
    for (int k = 0; k < KC_; k += 32) {
      v8h alo = *(const v8h*)(aLds + k);
      v8h ahi = *(const v8h*)(aLds + k + 16);
      v16h a = __builtin_shufflevector(alo, ahi,
                                       0,1,2,3,4,5,6,7,8,9,10,11,12,13,14,15);
#pragma unroll
      for (int ns = 0; ns < 8; ++ns) {
        const _Float16* bp = bBase + (size_t)ns * 16 * K_ + c * KC_ + k;
        v8h blo = *(const v8h*)(bp);
        v8h bhi = *(const v8h*)(bp + 16);
        v16h bfrag = __builtin_shufflevector(blo, bhi,
                                             0,1,2,3,4,5,6,7,8,9,10,11,12,13,14,15);
        acc[ns] = __builtin_amdgcn_wmma_f32_16x16x32_f16(
            false, a, false, bfrag, (short)0, acc[ns], false, false);
      }
    }
    __syncthreads();
  }

  // ---------- epilogue: per-wave 16x16 transpose in LDS, coalesced stores ----
  // C/D layout: lane l16 = N column, vgpr r = M row (m = r + 8*hi16).
  // Wave-private region; DS ops are in-order within a wave -> no barriers.
  float* treg = (float*)(&lds[0][0]) + (size_t)wave * 16 * TROW_;
  const int mq = (lane & 3) * 4;                       // m sub-column (x4)
  const int nq = lane >> 2;                            // 0..7 n row group
#pragma unroll
  for (int ns = 0; ns < 8; ++ns) {
    int n_w = nh * 128 + ns * 16 + l16;
    float bv = bias[n_w];
    v4f lo4 = {acc[ns][0] + bv, acc[ns][1] + bv, acc[ns][2] + bv, acc[ns][3] + bv};
    v4f hi4 = {acc[ns][4] + bv, acc[ns][5] + bv, acc[ns][6] + bv, acc[ns][7] + bv};
    *(v4f*)(treg + l16 * TROW_ + hi16 * 8)     = lo4;  // m cols hi16*8 .. +3
    *(v4f*)(treg + l16 * TROW_ + hi16 * 8 + 4) = hi4;  // m cols hi16*8+4 .. +7
#pragma unroll
    for (int it = 0; it < 2; ++it) {
      int n_local = it * 8 + nq;
      int n = nh * 128 + ns * 16 + n_local;
      v4f v = *(const v4f*)(treg + n_local * TROW_ + mq);
      int m  = mBase + r0 + mq;                        // 4-aligned run, no batch straddle
      int bb = m / HW_;
      int hw = m - bb * HW_;
      *(v4f*)(out + (size_t)bb * (COUT_ * HW_) + (size_t)n * HW_ + hw) = v;
    }
  }
}

// ---------------------------------------------------------------------------
// Host launcher. Inputs: x, flow, combinator, bias (fp32). Output fp32.
// Workspace: [xt fp16 6.25MB | comb fp16 0.5MB | tap table 200KB]
// ---------------------------------------------------------------------------
extern "C" void kernel_launch(void* const* d_in, const int* in_sizes, int n_in,
                              void* d_out, int out_size, void* d_ws, size_t ws_size,
                              hipStream_t stream) {
  const float* x    = (const float*)d_in[0];
  const float* flow = (const float*)d_in[1];
  const float* comb = (const float*)d_in[2];
  const float* bias = (const float*)d_in[3];
  float* out = (float*)d_out;

  char* ws = (char*)d_ws;
  const size_t xtBytes   = (size_t)M_ * CIN_ * sizeof(_Float16);  // 6,553,600
  const size_t combBytes = (size_t)N_ * K_ * sizeof(_Float16);    //   524,288
  _Float16* xt     = (_Float16*)ws;
  _Float16* comb_p = (_Float16*)(ws + xtBytes);
  Tap*      table  = (Tap*)(ws + xtBytes + combBytes);

  flow_table_kernel<<<(NF_ * HW_ + 255) / 256, 256, 0, stream>>>(flow, table);
  prep_xt_kernel   <<<(M_ + 255) / 256, 256, 0, stream>>>(x, xt);
  cvt_comb_kernel  <<<(N_ * K_ + 255) / 256, 256, 0, stream>>>(comb, comb_p);
  fused_warp_gemm_kernel<<<M_ / TM_, 256, 0, stream>>>(xt, table, comb_p, bias, out);
}
